// MaskedItemToInterestAggregation_8237747273810
// MI455X (gfx1250) — compile-verified, matrix operand order checked by reading, not run on hardware
//
#include <hip/hip_runtime.h>
#include <hip/hip_bf16.h>

#define BB  512
#define SS  200
#define HH  256
#define KK  16
#define SP  224          // S padded to multiple of 32 (WMMA K-step)
#define XSTR 264         // LDS row stride for X/h1 buffers (bf16 elems, 16B aligned, bank-skewed)
#define TSTR 224         // LDS row stride for X^T (s elems)
#define DPW  112         // packed-D words per k row (SP/2)

#define OFF_B   (SP * XSTR * 2)                 // 118272
#define OFF_D   (2 * SP * XSTR * 2)             // 236544
#define OFF_DP  (OFF_D + SP * 16 * 4)           // 250880
#define OFF_STG (OFF_DP + 16 * DPW * 4)         // 258048
#define SMEM_TOTAL (OFF_STG + 8 * 16 * 32 * 2)  // 266240

typedef __attribute__((ext_vector_type(16))) __bf16 v16bf;
typedef __attribute__((ext_vector_type(8)))  float  v8f;

union ABFrag { v16bf v; uint4 q[2]; };

__device__ __forceinline__ unsigned short f2bf(float f) {
  unsigned int u = __float_as_uint(f);
  u += 0x7FFFu + ((u >> 16) & 1u);            // round-to-nearest-even
  return (unsigned short)(u >> 16);
}

__device__ __forceinline__ v8f wmma_bf16(v16bf a, v16bf b, v8f c) {
  return __builtin_amdgcn_wmma_f32_16x16x32_bf16(false, a, false, b, (short)0, c,
                                                 false, false);
}

// ---------------------------------------------------------------------------
// Pre-pass: pack W1, W2 (256x256) and W3 (256x16) fp32 row-major -> bf16,
// column-major with K-pairs packed into 32-bit words:
//   Wp[n*128 + j] = { bf16(W[2j][n]), bf16(W[2j+1][n]) }
// so WMMA B-fragments become two 16B loads.
// ---------------------------------------------------------------------------
__global__ void pack_weights(const float* __restrict__ W1,
                             const float* __restrict__ W2,
                             const float* __restrict__ W3,
                             unsigned int* __restrict__ ws) {
  int idx = blockIdx.x * blockDim.x + threadIdx.x;
  if (idx < 32768) {                          // W1p
    int n = idx >> 7, j = idx & 127;
    unsigned int lo = f2bf(W1[(2 * j) * 256 + n]);
    unsigned int hi = f2bf(W1[(2 * j + 1) * 256 + n]);
    ws[idx] = lo | (hi << 16);
  } else if (idx < 65536) {                   // W2p
    int t = idx - 32768;
    int n = t >> 7, j = t & 127;
    unsigned int lo = f2bf(W2[(2 * j) * 256 + n]);
    unsigned int hi = f2bf(W2[(2 * j + 1) * 256 + n]);
    ws[idx] = lo | (hi << 16);
  } else if (idx < 67584) {                   // W3p (n = k in 0..15)
    int t = idx - 65536;
    int n = t >> 7, j = t & 127;
    unsigned int lo = f2bf(W3[(2 * j) * 16 + n]);
    unsigned int hi = f2bf(W3[(2 * j + 1) * 16 + n]);
    ws[idx] = lo | (hi << 16);
  }
}

// ---------------------------------------------------------------------------
// One M-chunk of a 256-wide GEMM over MC s-tiles and this wave's 2 column
// strips. FUSE=true additionally projects the (bias+ReLU'd) h2 tile through
// W3 and accumulates partial D tiles into LDS with ds_add_f32.
// ---------------------------------------------------------------------------
template <int MC, bool FUSE>
__device__ __forceinline__ void gemm_chunk(int m0,
                                           const unsigned short* __restrict__ src,
                                           unsigned short* __restrict__ dst,
                                           const unsigned int* __restrict__ Wpk,
                                           float bias0, float bias1,
                                           ABFrag W3f,
                                           unsigned short* __restrict__ stg,
                                           float* __restrict__ Dls,
                                           int wv, int l16, int kb) {
  const v8f zero = {0.f, 0.f, 0.f, 0.f, 0.f, 0.f, 0.f, 0.f};
  v8f acc[MC][2];
#pragma unroll
  for (int mm = 0; mm < MC; ++mm) { acc[mm][0] = zero; acc[mm][1] = zero; }

  for (int k = 0; k < 8; ++k) {               // K = 256 in steps of 32
    ABFrag Bf0, Bf1;
    const uint4* p0 = (const uint4*)(Wpk + (wv * 32 + l16) * 128 + k * 16 + (kb >> 1));
    Bf0.q[0] = p0[0]; Bf0.q[1] = p0[2];
    const uint4* p1 = (const uint4*)(Wpk + (wv * 32 + 16 + l16) * 128 + k * 16 + (kb >> 1));
    Bf1.q[0] = p1[0]; Bf1.q[1] = p1[2];
    __builtin_prefetch((const void*)(p0 + 4), 0, 0);   // next K-step of weights
#pragma unroll
    for (int mm = 0; mm < MC; ++mm) {
      const uint4* ap = (const uint4*)(src + ((m0 + mm) * 16 + l16) * XSTR + k * 32 + kb);
      ABFrag Af; Af.q[0] = ap[0]; Af.q[1] = ap[2];
      acc[mm][0] = wmma_bf16(Af.v, Bf0.v, acc[mm][0]);
      acc[mm][1] = wmma_bf16(Af.v, Bf1.v, acc[mm][1]);
    }
  }

#pragma unroll
  for (int mm = 0; mm < MC; ++mm) {
    if (!FUSE) {
#pragma unroll
      for (int i = 0; i < 2; ++i) {
        float bias = i ? bias1 : bias0;
        int col = wv * 32 + i * 16 + l16;
#pragma unroll
        for (int r = 0; r < 8; ++r) {
          float v = fmaxf(acc[mm][i][r] + bias, 0.0f);
          dst[((m0 + mm) * 16 + r + kb) * XSTR + col] = f2bf(v);
        }
      }
    } else {
      // stage h2 tile (16 x 32, this wave's column strip) as bf16 A-fragment
#pragma unroll
      for (int i = 0; i < 2; ++i) {
        float bias = i ? bias1 : bias0;
#pragma unroll
        for (int r = 0; r < 8; ++r) {
          float v = fmaxf(acc[mm][i][r] + bias, 0.0f);
          stg[(r + kb) * 32 + i * 16 + l16] = f2bf(v);
        }
      }
      const uint4* sp = (const uint4*)(stg + l16 * 32 + kb);
      ABFrag Af; Af.q[0] = sp[0]; Af.q[1] = sp[2];
      const v8f z = {0.f, 0.f, 0.f, 0.f, 0.f, 0.f, 0.f, 0.f};
      v8f d = wmma_bf16(Af.v, W3f.v, z);
#pragma unroll
      for (int r = 0; r < 8; ++r)
        atomicAdd(&Dls[((m0 + mm) * 16 + r + kb) * 16 + l16], d[r]);  // ds_add_f32
    }
  }
}

// ---------------------------------------------------------------------------
// Fused kernel: one workgroup per batch element.
// ---------------------------------------------------------------------------
__global__ void __launch_bounds__(256)
fused_kernel(const float* __restrict__ X, const unsigned char* __restrict__ mask,
             const float* __restrict__ timef, const float* __restrict__ b1,
             const float* __restrict__ b2, const unsigned int* __restrict__ Wp,
             float* __restrict__ out) {
  extern __shared__ unsigned char smem[];
  unsigned short* bufA  = (unsigned short*)smem;              // X  bf16 [224][264]
  unsigned short* bufB  = (unsigned short*)(smem + OFF_B);    // h1 bf16, later X^T
  float*          Dls   = (float*)(smem + OFF_D);             // D  fp32 [224][16]
  unsigned int*   Dp    = (unsigned int*)(smem + OFF_DP);     // packed D [16][112]
  unsigned short* stage = (unsigned short*)(smem + OFF_STG);  // per-wave 16x32

  const int b   = blockIdx.x;
  const int tid = threadIdx.x;
  const int wv  = tid >> 5;
  const int lane = tid & 31;
  const int l16 = lane & 15;
  const int hi  = lane >> 4;
  const int kb  = hi << 3;                    // 0 or 8 (K sub-group)

  const unsigned int* W1p = Wp;
  const unsigned int* W2p = Wp + 32768;
  const unsigned int* W3p = Wp + 65536;

  // ---- Phase 0: zero D accumulator, stage X[b] -> LDS bf16 (zero-pad rows)
  for (int i = tid; i < SP * 16; i += 256) Dls[i] = 0.0f;
  const float* Xb = X + (size_t)b * SS * HH;
  for (int i = tid; i < SP * HH; i += 256) {
    int s = i >> 8, h = i & 255;
    float v = (s < SS) ? Xb[s * HH + h] : 0.0f;
    bufA[s * XSTR + h] = f2bf(v);
  }
  __syncthreads();

  // ---- GEMM1: h1 = relu(X @ W1 + b1) -> bufB (bf16)
  {
    ABFrag dummy; dummy.q[0] = uint4{0,0,0,0}; dummy.q[1] = uint4{0,0,0,0};
    float bias0 = b1[wv * 32 + l16];
    float bias1 = b1[wv * 32 + 16 + l16];
    gemm_chunk<4, false>(0,  bufA, bufB, W1p, bias0, bias1, dummy, nullptr, nullptr, wv, l16, kb);
    gemm_chunk<4, false>(4,  bufA, bufB, W1p, bias0, bias1, dummy, nullptr, nullptr, wv, l16, kb);
    gemm_chunk<4, false>(8,  bufA, bufB, W1p, bias0, bias1, dummy, nullptr, nullptr, wv, l16, kb);
    gemm_chunk<2, false>(12, bufA, bufB, W1p, bias0, bias1, dummy, nullptr, nullptr, wv, l16, kb);
  }
  __syncthreads();

  // ---- GEMM2 fused with W3 projection: D += relu(h1 @ W2 + b2) @ W3
  {
    float bias0 = b2[wv * 32 + l16];
    float bias1 = b2[wv * 32 + 16 + l16];
    ABFrag W3f;                               // K-slice = this wave's 32 h-columns
    const uint4* p3 = (const uint4*)(W3p + l16 * 128 + wv * 16 + (kb >> 1));
    W3f.q[0] = p3[0]; W3f.q[1] = p3[2];
    unsigned short* stg = stage + wv * (16 * 32);
    gemm_chunk<4, true>(0,  bufB, nullptr, W2p, bias0, bias1, W3f, stg, Dls, wv, l16, kb);
    gemm_chunk<4, true>(4,  bufB, nullptr, W2p, bias0, bias1, W3f, stg, Dls, wv, l16, kb);
    gemm_chunk<4, true>(8,  bufB, nullptr, W2p, bias0, bias1, W3f, stg, Dls, wv, l16, kb);
    gemm_chunk<2, true>(12, bufB, nullptr, W2p, bias0, bias1, W3f, stg, Dls, wv, l16, kb);
  }
  __syncthreads();

  // ---- Row softmax over K=16, mask -> -10000, /sqrt(16)
  if (tid < SS) {
    int s = tid;
    float v[16];
    const float4* dr = (const float4*)(Dls + s * 16);
    float4 a0 = dr[0], a1 = dr[1], a2 = dr[2], a3 = dr[3];
    v[0]=a0.x; v[1]=a0.y; v[2]=a0.z; v[3]=a0.w;  v[4]=a1.x; v[5]=a1.y; v[6]=a1.z; v[7]=a1.w;
    v[8]=a2.x; v[9]=a2.y; v[10]=a2.z; v[11]=a2.w; v[12]=a3.x; v[13]=a3.y; v[14]=a3.z; v[15]=a3.w;
    float mx = v[0];
#pragma unroll
    for (int k = 1; k < 16; ++k) mx = fmaxf(mx, v[k]);
    float sum = 0.f;
#pragma unroll
    for (int k = 0; k < 16; ++k) { v[k] = __expf(v[k] - mx); sum += v[k]; }
    float inv = 1.0f / sum;
    unsigned char mk = mask[(size_t)b * SS + s];
#pragma unroll
    for (int k = 0; k < 16; ++k) {
      float p = mk ? (v[k] * inv) : -10000.0f;
      Dls[s * 16 + k] = p * 0.25f;
    }
  }
  __syncthreads();

  // ---- Column softmax over S (16 lane-groups of 16: group = column k)
  {
    int g = wv * 2 + hi;                      // column k handled by this lane-group
    float mx = -3.0e38f;
    for (int s = l16; s < SS; s += 16) mx = fmaxf(mx, Dls[s * 16 + g]);
#pragma unroll
    for (int off = 1; off < 16; off <<= 1) mx = fmaxf(mx, __shfl_xor(mx, off, 16));
    float sum = 0.f;
    for (int s = l16; s < SS; s += 16) sum += __expf(Dls[s * 16 + g] - mx);
#pragma unroll
    for (int off = 1; off < 16; off <<= 1) sum += __shfl_xor(sum, off, 16);
    float inv = 1.0f / sum;
    for (int s = l16; s < SS; s += 16) {
      float p = __expf(Dls[s * 16 + g] - mx) * inv * timef[(size_t)b * SS + s];
      Dls[s * 16 + g] = p;
    }
  }
  __syncthreads();

  // ---- Pack D -> bf16 K-pair words (zero pad s>=200); reload X as X^T bf16
  for (int i = tid; i < 16 * DPW; i += 256) {
    int k = i / DPW, j = i - k * DPW;
    int s0 = 2 * j;
    float v0 = (s0     < SS) ? Dls[s0 * 16 + k]       : 0.0f;
    float v1 = (s0 + 1 < SS) ? Dls[(s0 + 1) * 16 + k] : 0.0f;
    Dp[k * DPW + j] = (unsigned int)f2bf(v0) | ((unsigned int)f2bf(v1) << 16);
  }
  for (int i = tid; i < SS * HH; i += 256) {  // coalesced global read, transposed LDS write
    int s = i >> 8, h = i & 255;
    bufB[h * TSTR + s] = f2bf(Xb[i]);
  }
  for (int i = tid; i < HH * (SP - SS); i += 256) {
    int h = i / (SP - SS), s = SS + i - h * (SP - SS);
    bufB[h * TSTR + s] = 0;                   // zero pads (WMMA NaN safety)
  }
  __syncthreads();

  // ---- Final: out[b] = D^T (16 x S) @ X (S x 256), Kdim = s
  {
    const v8f zero = {0.f, 0.f, 0.f, 0.f, 0.f, 0.f, 0.f, 0.f};
    v8f acc0 = zero, acc1 = zero;
    for (int ks = 0; ks < 7; ++ks) {          // 7 * 32 = 224 padded s
      const uint4* ap = (const uint4*)(Dp + l16 * DPW + ks * 16 + (kb >> 1));
      ABFrag Af; Af.q[0] = ap[0]; Af.q[1] = ap[2];
      const uint4* bp0 = (const uint4*)(bufB + (wv * 32 + l16) * TSTR + ks * 32 + kb);
      ABFrag Bf0; Bf0.q[0] = bp0[0]; Bf0.q[1] = bp0[2];
      const uint4* bp1 = (const uint4*)(bufB + (wv * 32 + 16 + l16) * TSTR + ks * 32 + kb);
      ABFrag Bf1; Bf1.q[0] = bp1[0]; Bf1.q[1] = bp1[2];
      acc0 = wmma_bf16(Af.v, Bf0.v, acc0);
      acc1 = wmma_bf16(Af.v, Bf1.v, acc1);
    }
    float* ob = out + (size_t)b * KK * HH;
#pragma unroll
    for (int i = 0; i < 2; ++i) {
      v8f a = i ? acc1 : acc0;
      int h = wv * 32 + i * 16 + l16;
#pragma unroll
      for (int r = 0; r < 8; ++r) {
        int k = r + kb;                       // C layout: M = r (+8 for upper lanes)
        ob[k * HH + h] = a[r];
      }
    }
  }
}

extern "C" void kernel_launch(void* const* d_in, const int* in_sizes, int n_in,
                              void* d_out, int out_size, void* d_ws, size_t ws_size,
                              hipStream_t stream) {
  const float*         X     = (const float*)d_in[0];
  const unsigned char* mask  = (const unsigned char*)d_in[1];
  const float*         timef = (const float*)d_in[2];
  const float*         W1    = (const float*)d_in[3];
  const float*         b1    = (const float*)d_in[4];
  const float*         W2    = (const float*)d_in[5];
  const float*         b2    = (const float*)d_in[6];
  const float*         W3    = (const float*)d_in[7];
  float*               out   = (float*)d_out;
  unsigned int*        Wp    = (unsigned int*)d_ws;   // 270,336 B used

  pack_weights<<<264, 256, 0, stream>>>(W1, W2, W3, Wp);

  (void)hipFuncSetAttribute((const void*)fused_kernel,
                            hipFuncAttributeMaxDynamicSharedMemorySize, SMEM_TOTAL);
  fused_kernel<<<BB, 256, SMEM_TOTAL, stream>>>(X, mask, timef, b1, b2, Wp, out);
}
